// Attention_37984690766438
// MI455X (gfx1250) — matrix operand order repaired and checked by reading, not compiled, not run on hardware
//
#include <hip/hip_runtime.h>
#include <stdint.h>

#define SEQ   4096
#define DIM   1024
#define HEADS 16
#define HD    64
#define QKVN  (3 * DIM)

typedef __attribute__((ext_vector_type(16))) __bf16 v16bf;
typedef __attribute__((ext_vector_type(8)))  __bf16 v8bf;
typedef __attribute__((ext_vector_type(8)))  float  v8f;

__device__ __forceinline__ uint16_t f32_to_bf16(float f) {
    uint32_t u = __builtin_bit_cast(uint32_t, f);
    u += 0x7FFFu + ((u >> 16) & 1u);   // round-to-nearest-even
    return (uint16_t)(u >> 16);
}

__device__ __forceinline__ v16bf cat8(v8bf lo, v8bf hi) {
    return __builtin_shufflevector(lo, hi, 0,1,2,3,4,5,6,7,8,9,10,11,12,13,14,15);
}
__device__ __forceinline__ v16bf load_v16(const uint16_t* p) { return *(const v16bf*)p; }
__device__ __forceinline__ v8bf  load_v8 (const uint16_t* p) { return *(const v8bf*)p; }

__device__ __forceinline__ v8f wmma_bf16(v16bf a, v16bf b, v8f c) {
    return __builtin_amdgcn_wmma_f32_16x16x32_bf16(false, a, false, b, (short)0, c,
                                                   false, false);
}

// generic LDS pointer -> wave-relative LDS byte address (low 32 bits of the
// shared aperture address are the LDS offset per CDNA5 flat addressing rules)
__device__ __forceinline__ uint32_t ldsoff(const void* p) {
    return (uint32_t)(uintptr_t)p;
}

// async copy 16B global -> LDS, tracked by ASYNCcnt (CDNA5 ISA)
__device__ __forceinline__ void async_copy16(uint32_t lds, const void* g) {
    asm volatile("global_load_async_to_lds_b128 %0, %1, off"
                 :: "v"(lds), "v"(g) : "memory");
}
__device__ __forceinline__ void wait_async_le2() {
    asm volatile("s_wait_asynccnt 0x2" ::: "memory");
}
__device__ __forceinline__ void wait_async_le1() {
    asm volatile("s_wait_asynccnt 0x1" ::: "memory");
}
__device__ __forceinline__ void wait_async_0() {
    asm volatile("s_wait_asynccnt 0x0" ::: "memory");
}

// per-wave A operand (16x32 bf16, K-contiguous rows) k = 16*lh + s (contiguous), same order as B
__device__ __forceinline__ v16bf loadA(const uint16_t* arow, int kk, int lh) {
    return load_v16(arow + kk + lh * 16);
}

// ---------------------------------------------------------------- f32 -> bf16
__global__ void cvt_kernel(const float* __restrict__ in, uint16_t* __restrict__ out, int n) {
    int i = blockIdx.x * blockDim.x + threadIdx.x;
    if (i < n) out[i] = f32_to_bf16(in[i]);
}

// ------------------------------------------------------------- QKV projection
// Block = 8 waves covering 128(M) x 64(N); B tile (64x32) async-staged in LDS,
// double buffered; scatter into Q[h][s][d], K[h][s][d], Vt[h][d][s] (bf16).
__global__ __launch_bounds__(256) void qkv_gemm_kernel(
    const uint16_t* __restrict__ Xb, const uint16_t* __restrict__ Wb,
    const float* __restrict__ bias,
    uint16_t* __restrict__ Q, uint16_t* __restrict__ K, uint16_t* __restrict__ Vt)
{
    __shared__ __align__(64) uint16_t sB[2][64 * 32];
    const int tid  = threadIdx.x;
    const int lane = tid & 31, wave = tid >> 5;
    const int nTiles = QKVN / 64;                   // 48
    const int mBase = (blockIdx.x / nTiles) * 128 + wave * 16;
    const int nBase = (blockIdx.x % nTiles) * 64;
    const int l16 = lane & 15, lh = lane >> 4;
    const int n_i = tid >> 2, c_i = tid & 3;        // 256 threads stage 64x32 tile

    const uint16_t* arow = Xb + (size_t)(mBase + l16) * DIM;
    const v8f vzero = {0.f,0.f,0.f,0.f,0.f,0.f,0.f,0.f};
    v8f acc[4] = {vzero, vzero, vzero, vzero};

    // prologue: stage first B tile
    async_copy16(ldsoff(&sB[0][n_i * 32 + c_i * 8]),
                 Wb + (size_t)(nBase + n_i) * DIM + 0 + c_i * 8);
    v16bf A = loadA(arow, 0, lh);

    for (int kk = 0; kk < DIM; kk += 32) {
        const int b = (kk >> 5) & 1;
        v16bf An = A;
        if (kk + 32 < DIM) {
            async_copy16(ldsoff(&sB[b ^ 1][n_i * 32 + c_i * 8]),
                         Wb + (size_t)(nBase + n_i) * DIM + (kk + 32) + c_i * 8);
            An = loadA(arow, kk + 32, lh);
            wait_async_le1();                       // current tile landed (ours)
        } else {
            wait_async_0();
        }
        __syncthreads();                            // everyone's portion landed
        #pragma unroll
        for (int t = 0; t < 4; ++t)
            acc[t] = wmma_bf16(A, load_v16(&sB[b][(t * 16 + l16) * 32 + lh * 16]),
                               acc[t]);
        __syncthreads();                            // reads done before overwrite
        A = An;
    }

    const int section = nBase / DIM;                // 0:q 1:k 2:v
    const int head = (nBase % DIM) / HD;
    #pragma unroll
    for (int t = 0; t < 4; ++t) {
        const int n = nBase + t * 16 + l16;
        const int d = t * 16 + l16;
        const float bn = bias[n];
        #pragma unroll
        for (int g = 0; g < 8; ++g) {
            const float v = acc[t][g] + bn;
            const int q = mBase + g + 8 * lh;       // C layout: row = g + 8*(lane/16)
            const uint16_t hv = f32_to_bf16(v);
            if (section == 0)      Q [((size_t)head * SEQ + q) * HD + d] = hv;
            else if (section == 1) K [((size_t)head * SEQ + q) * HD + d] = hv;
            else                   Vt[((size_t)head * HD + d) * SEQ + q] = hv;
        }
    }
}

// -------------------------------------------------- flash attention per head
// Block = 8 waves, all on the SAME head (8 query tiles of 16). K/V tiles for
// each 32-key step are shared -> async-staged to LDS once per block, double
// buffered. Online softmax per wave; P relayout via per-wave LDS tile.
__global__ __launch_bounds__(256) void attn_kernel(
    const uint16_t* __restrict__ Q, const uint16_t* __restrict__ Km,
    const uint16_t* __restrict__ Vt, uint16_t* __restrict__ O)
{
    __shared__ __align__(64) uint16_t sK[2][32 * 64];   // [key-local][d]
    __shared__ __align__(64) uint16_t sV[2][64 * 32];   // [d][key-local]
    __shared__ __align__(64) uint16_t sP[8][16 * 32];   // per-wave P tile
    const int tid  = threadIdx.x;
    const int lane = tid & 31, wave = tid >> 5;
    const int w = blockIdx.x * 8 + wave;
    const int head = w >> 8;                        // 256 q-tiles per head
    const int qBase = (w & 255) * 16;
    const int l16 = lane & 15, lh = lane >> 4;
    uint16_t* pl = sP[wave];

    const uint16_t* Kbase = Km + (size_t)head * SEQ * HD;
    const uint16_t* Vbase = Vt + (size_t)head * HD * SEQ;
    const int d_i = tid >> 2, c_i = tid & 3;

    // Q tile as two A-operands (d = 0..31 and 32..63), loaded once
    const uint16_t* qrow = Q + ((size_t)head * SEQ + qBase + l16) * HD;
    v16bf Aq[2];
    #pragma unroll
    for (int dh = 0; dh < 2; ++dh)
        Aq[dh] = load_v16(qrow + dh * 32 + lh * 16);

    // prologue: stage K/V tiles for kk = 0 (K tile is 4KB contiguous)
    async_copy16(ldsoff(&sK[0][tid * 8]), Kbase + (size_t)0 * HD + tid * 8);
    async_copy16(ldsoff(&sV[0][d_i * 32 + c_i * 8]),
                 Vbase + (size_t)d_i * SEQ + 0 + c_i * 8);

    const v8f vzero = {0.f,0.f,0.f,0.f,0.f,0.f,0.f,0.f};
    v8f o[4] = {vzero, vzero, vzero, vzero};
    float m[8], l[8];
    #pragma unroll
    for (int g = 0; g < 8; ++g) { m[g] = -1e30f; l[g] = 0.f; }

    const float LOG2E = 1.4426950408889634f;
    const float SCALE = 0.125f;                     // 1/sqrt(64)

    for (int kk = 0; kk < SEQ; kk += 32) {
        const int b = (kk >> 5) & 1;
        if (kk + 32 < SEQ) {
            async_copy16(ldsoff(&sK[b ^ 1][tid * 8]),
                         Kbase + (size_t)(kk + 32) * HD + tid * 8);
            async_copy16(ldsoff(&sV[b ^ 1][d_i * 32 + c_i * 8]),
                         Vbase + (size_t)d_i * SEQ + (kk + 32) + c_i * 8);
            wait_async_le2();                       // current pair landed (ours)
        } else {
            wait_async_0();
        }
        __syncthreads();

        // S = Q K^T : two 16-key N tiles, K-dim 64 = two WMMAs each (from LDS)
        const uint16_t* kt0 = &sK[b][l16 * 64 + lh * 16];
        v8f s0 = vzero, s1 = vzero;
        s0 = wmma_bf16(Aq[0], load_v16(kt0),                s0);
        s0 = wmma_bf16(Aq[1], load_v16(kt0 + 32),           s0);
        s1 = wmma_bf16(Aq[0], load_v16(kt0 + 16 * 64),      s1);
        s1 = wmma_bf16(Aq[1], load_v16(kt0 + 16 * 64 + 32), s1);

        // online softmax; a row's 16 cols live in one 16-lane group
        #pragma unroll
        for (int g = 0; g < 8; ++g) {
            float a = s0[g] * SCALE, bb = s1[g] * SCALE;
            float rm = fmaxf(a, bb);
            rm = fmaxf(rm, __shfl_xor(rm, 1, 32));
            rm = fmaxf(rm, __shfl_xor(rm, 2, 32));
            rm = fmaxf(rm, __shfl_xor(rm, 4, 32));
            rm = fmaxf(rm, __shfl_xor(rm, 8, 32));
            const float mn = fmaxf(m[g], rm);
            const float alpha = exp2f((m[g] - mn) * LOG2E);
            const float p0 = exp2f((a - mn) * LOG2E);
            const float p1 = exp2f((bb - mn) * LOG2E);
            float rs = p0 + p1;
            rs += __shfl_xor(rs, 1, 32);
            rs += __shfl_xor(rs, 2, 32);
            rs += __shfl_xor(rs, 4, 32);
            rs += __shfl_xor(rs, 8, 32);
            l[g] = l[g] * alpha + rs;
            m[g] = mn;
            #pragma unroll
            for (int t = 0; t < 4; ++t) o[t][g] *= alpha;
            const int r = g + 8 * lh;
            pl[r * 32 + l16]      = f32_to_bf16(p0);
            pl[r * 32 + l16 + 16] = f32_to_bf16(p1);
        }
        asm volatile("s_wait_dscnt 0" ::: "memory");   // cross-lane LDS RAW

        // P in A-operand layout; O += P @ V (Vt rows in LDS are B columns)
        v16bf Ap = load_v16(pl + l16 * 32 + lh * 16);
        #pragma unroll
        for (int t = 0; t < 4; ++t)
            o[t] = wmma_bf16(Ap, load_v16(&sV[b][(t * 16 + l16) * 32 + lh * 16]),
                             o[t]);
        __syncthreads();                            // reads done before overwrite
    }

    #pragma unroll
    for (int t = 0; t < 4; ++t) {
        #pragma unroll
        for (int g = 0; g < 8; ++g) {
            const float val = o[t][g] / l[g];
            const int q = qBase + g + 8 * lh;
            O[(size_t)q * DIM + head * HD + t * 16 + l16] = f32_to_bf16(val);
        }
    }
}

// ----------------------------------------------------------- output projection
// out[4096,1024] = Ab @ Wb^T + bias (f32); same block-cooperative scheme.
__global__ __launch_bounds__(256) void proj_gemm_kernel(
    const uint16_t* __restrict__ Ab, const uint16_t* __restrict__ Wb,
    const float* __restrict__ bias, float* __restrict__ out)
{
    __shared__ __align__(64) uint16_t sB[2][64 * 32];
    const int tid  = threadIdx.x;
    const int lane = tid & 31, wave = tid >> 5;
    const int nTiles = DIM / 64;                    // 16
    const int mBase = (blockIdx.x / nTiles) * 128 + wave * 16;
    const int nBase = (blockIdx.x % nTiles) * 64;
    const int l16 = lane & 15, lh = lane >> 4;
    const int n_i = tid >> 2, c_i = tid & 3;

    const uint16_t* arow = Ab + (size_t)(mBase + l16) * DIM;
    const v8f vzero = {0.f,0.f,0.f,0.f,0.f,0.f,0.f,0.f};
    v8f acc[4] = {vzero, vzero, vzero, vzero};

    async_copy16(ldsoff(&sB[0][n_i * 32 + c_i * 8]),
                 Wb + (size_t)(nBase + n_i) * DIM + 0 + c_i * 8);
    v16bf A = loadA(arow, 0, lh);

    for (int kk = 0; kk < DIM; kk += 32) {
        const int b = (kk >> 5) & 1;
        v16bf An = A;
        if (kk + 32 < DIM) {
            async_copy16(ldsoff(&sB[b ^ 1][n_i * 32 + c_i * 8]),
                         Wb + (size_t)(nBase + n_i) * DIM + (kk + 32) + c_i * 8);
            An = loadA(arow, kk + 32, lh);
            wait_async_le1();
        } else {
            wait_async_0();
        }
        __syncthreads();
        #pragma unroll
        for (int t = 0; t < 4; ++t)
            acc[t] = wmma_bf16(A, load_v16(&sB[b][(t * 16 + l16) * 32 + lh * 16]),
                               acc[t]);
        __syncthreads();
        A = An;
    }

    #pragma unroll
    for (int t = 0; t < 4; ++t) {
        const int n = nBase + t * 16 + l16;
        const float bn = bias[n];
        #pragma unroll
        for (int g = 0; g < 8; ++g) {
            const int q = mBase + g + 8 * lh;
            out[(size_t)q * DIM + n] = acc[t][g] + bn;
        }
    }
}

// -----------------------------------------------------------------------------
extern "C" void kernel_launch(void* const* d_in, const int* in_sizes, int n_in,
                              void* d_out, int out_size, void* d_ws, size_t ws_size,
                              hipStream_t stream) {
    const float* x     = (const float*)d_in[0];
    const float* Wqkv  = (const float*)d_in[1];
    const float* bqkv  = (const float*)d_in[2];
    const float* Wproj = (const float*)d_in[3];
    const float* bproj = (const float*)d_in[4];
    float* out = (float*)d_out;

    uint8_t* ws = (uint8_t*)d_ws;
    size_t off = 0;
    auto take = [&](size_t elems) -> uint16_t* {
        uint16_t* p = (uint16_t*)(ws + off);
        off += (elems * sizeof(uint16_t) + 255) & ~(size_t)255;
        return p;
    };
    uint16_t* Xb     = take((size_t)SEQ * DIM);
    uint16_t* Wqkvb  = take((size_t)QKVN * DIM);
    uint16_t* Wprojb = take((size_t)DIM * DIM);
    uint16_t* Qb     = take((size_t)SEQ * DIM);
    uint16_t* Kb     = take((size_t)SEQ * DIM);
    uint16_t* Vtb    = take((size_t)SEQ * DIM);
    uint16_t* Ob     = take((size_t)SEQ * DIM);

    const int CB = 256;
    cvt_kernel<<<(SEQ * DIM + CB - 1) / CB, CB, 0, stream>>>(x, Xb, SEQ * DIM);
    cvt_kernel<<<(QKVN * DIM + CB - 1) / CB, CB, 0, stream>>>(Wqkv, Wqkvb, QKVN * DIM);
    cvt_kernel<<<(DIM * DIM + CB - 1) / CB, CB, 0, stream>>>(Wproj, Wprojb, DIM * DIM);

    qkv_gemm_kernel<<<(SEQ / 128) * (QKVN / 64), 256, 0, stream>>>(
        Xb, Wqkvb, bqkv, Qb, Kb, Vtb);
    attn_kernel<<<HEADS * (SEQ / 16) / 8, 256, 0, stream>>>(Qb, Kb, Vtb, Ob);
    proj_gemm_kernel<<<(SEQ / 128) * (DIM / 64), 256, 0, stream>>>(
        Ob, Wprojb, bproj, out);
}
